// Decoder_42571715838073
// MI455X (gfx1250) — matrix operand order repaired and checked by reading, not compiled, not run on hardware
//
#include <hip/hip_runtime.h>
#include <hip/hip_bf16.h>
#include <math.h>

// ---------------------------------------------------------------------------
// Shapes (fixed by the reference)
// ---------------------------------------------------------------------------
#define VOCAB  50000
#define INPUT  512
#define HIDDEN 1024
#define BATCH  64
#define MAXLEN 256

typedef __attribute__((ext_vector_type(16))) __bf16 v16bf;
typedef __attribute__((ext_vector_type(8)))  float  v8f;

// fp32 -> bf16 via the compiler's native conversion (RNE). On gfx1250 this
// should lower to hardware bf16 converts (v_cvt_pk_bf16_f32) instead of the
// 3-VALU-per-element integer emulation.
static __device__ __forceinline__ __bf16 f2bf(float f) {
    return (__bf16)f;
}

// ---------------------------------------------------------------------------
// Kernel 1: u2[k] = sum_h v[h] * attn_W[h, HIDDEN + k]   (k in [0,HIDDEN))
// (the hidden-state half of u cancels under softmax, so we only need cols H..2H)
// ---------------------------------------------------------------------------
__global__ __launch_bounds__(256) void k_attn_u2(const float* __restrict__ attn_W,
                                                 const float* __restrict__ v,
                                                 float* __restrict__ u2) {
    int k = blockIdx.x * 256 + threadIdx.x;           // grid = HIDDEN/256
    float acc = 0.f;
    for (int h = 0; h < HIDDEN; ++h)
        acc += v[h] * attn_W[(size_t)h * (2 * HIDDEN) + HIDDEN + k];
    u2[k] = acc;
}

// ---------------------------------------------------------------------------
// Kernel 2: per-batch attention scores, softmax over T, and context vector.
// One block per batch element b. 256 threads (T == 256).
// ---------------------------------------------------------------------------
__global__ __launch_bounds__(256) void k_attn_ctx(const float* __restrict__ enc,
                                                  const float* __restrict__ u2,
                                                  float* __restrict__ ctx) {
    __shared__ float s_u[HIDDEN];
    __shared__ float s_sc[MAXLEN];
    __shared__ float s_red[256];
    const int b = blockIdx.x, tid = threadIdx.x;

    for (int i = tid; i < HIDDEN; i += 256) s_u[i] = u2[i];
    __syncthreads();

    // score for timestep t = tid
    const float* e = enc + ((size_t)tid * BATCH + b) * HIDDEN;
    float s = 0.f;
    for (int h = 0; h < HIDDEN; h += 4) {
        float4 q = *reinterpret_cast<const float4*>(e + h);
        s += q.x * s_u[h] + q.y * s_u[h + 1] + q.z * s_u[h + 2] + q.w * s_u[h + 3];
    }
    s_sc[tid] = s; s_red[tid] = s; __syncthreads();
    for (int off = 128; off > 0; off >>= 1) {
        if (tid < off) s_red[tid] = fmaxf(s_red[tid], s_red[tid + off]);
        __syncthreads();
    }
    float m = s_red[0]; __syncthreads();
    float ex = __expf(s_sc[tid] - m);
    s_sc[tid] = ex; s_red[tid] = ex; __syncthreads();
    for (int off = 128; off > 0; off >>= 1) {
        if (tid < off) s_red[tid] += s_red[tid + off];
        __syncthreads();
    }
    const float inv = 1.0f / s_red[0];

    // context[b, h] = sum_t attn[t] * enc[t, b, h]; 4 h-values per thread
    float a0 = 0.f, a1 = 0.f, a2 = 0.f, a3 = 0.f;
    for (int t = 0; t < MAXLEN; ++t) {
        const float w = s_sc[t];
        const float* et = enc + ((size_t)t * BATCH + b) * HIDDEN + tid;
        a0 += w * et[0]; a1 += w * et[256]; a2 += w * et[512]; a3 += w * et[768];
    }
    float* c = ctx + (size_t)b * HIDDEN + tid;
    c[0] = a0 * inv; c[256] = a1 * inv; c[512] = a2 * inv; c[768] = a3 * inv;
}

// ---------------------------------------------------------------------------
// Kernel 3: x = [embedding[last_output[b]], context[b]]  -> [BATCH, INPUT+HIDDEN]
// ---------------------------------------------------------------------------
__global__ __launch_bounds__(256) void k_build_x(const int* __restrict__ last_out,
                                                 const float* __restrict__ emb,
                                                 const float* __restrict__ ctx,
                                                 float* __restrict__ x) {
    const int b = blockIdx.x, tid = threadIdx.x;
    const int tok = last_out[b];
    float* xb = x + (size_t)b * (INPUT + HIDDEN);
    for (int i = tid; i < INPUT; i += 256)  xb[i] = emb[(size_t)tok * INPUT + i];
    for (int i = tid; i < HIDDEN; i += 256) xb[INPUT + i] = ctx[(size_t)b * HIDDEN + i];
}

// ---------------------------------------------------------------------------
// Generic WMMA GEMM:  C[64, N] = A[64, K] @ W[N, K]^T + bias
// fp32 in memory, bf16 in the matrix core (v_wmma_f32_16x16x32_bf16).
// Block = 128 threads (4 waves); each wave owns one 16-column tile and all
// four 16-row tiles of M=64 (4 x v8f accumulators). W streamed exactly once.
// ---------------------------------------------------------------------------
__global__ __launch_bounds__(128) void k_gemm_bf16(const float* __restrict__ A,
                                                   const float* __restrict__ W,
                                                   const float* __restrict__ bias,
                                                   float* __restrict__ C,
                                                   int N, int K) {
    const int wave  = threadIdx.x >> 5;
    const int lane  = threadIdx.x & 31;
    const int ntile = blockIdx.x * 4 + wave;
    if (ntile * 16 >= N) return;                  // whole-wave exit: EXEC stays all-1s
    const int  n0   = ntile * 16;
    const int  col  = lane & 15;
    const bool hi   = lane >= 16;
    const int  aoff = hi ? 8 : 0;                 // A: lo lanes K {0..7,16..23}, hi {8..15,24..31}

    v8f acc[4] = {v8f{}, v8f{}, v8f{}, v8f{}};
    const float* wrow = W + (size_t)(n0 + col) * K + (hi ? 8 : 0); // B: lo lanes K {0..7,16..23}, hi {8..15,24..31}

    for (int k = 0; k < K; k += 32) {
        // ---- B tile (32x16 K-major), per-lane two 32B contiguous reads of out_W row ----
        v16bf bv;
        {
            const float4* p = reinterpret_cast<const float4*>(wrow + k);
            float4 q0 = p[0], q1 = p[1], q2 = p[4], q3 = p[5];
            bv[0]  = f2bf(q0.x); bv[1]  = f2bf(q0.y); bv[2]  = f2bf(q0.z); bv[3]  = f2bf(q0.w);
            bv[4]  = f2bf(q1.x); bv[5]  = f2bf(q1.y); bv[6]  = f2bf(q1.z); bv[7]  = f2bf(q1.w);
            bv[8]  = f2bf(q2.x); bv[9]  = f2bf(q2.y); bv[10] = f2bf(q2.z); bv[11] = f2bf(q2.w);
            bv[12] = f2bf(q3.x); bv[13] = f2bf(q3.y); bv[14] = f2bf(q3.z); bv[15] = f2bf(q3.w);
        }
        // ---- 4 A tiles (16x32 each) and 4 WMMAs ----
        #pragma unroll
        for (int mt = 0; mt < 4; ++mt) {
            const float* ar = A + (size_t)(mt * 16 + col) * K + k + aoff;
            float4 a0 = *reinterpret_cast<const float4*>(ar);
            float4 a1 = *reinterpret_cast<const float4*>(ar + 4);
            float4 a2 = *reinterpret_cast<const float4*>(ar + 16);
            float4 a3 = *reinterpret_cast<const float4*>(ar + 20);
            v16bf av;
            av[0]  = f2bf(a0.x); av[1]  = f2bf(a0.y); av[2]  = f2bf(a0.z); av[3]  = f2bf(a0.w);
            av[4]  = f2bf(a1.x); av[5]  = f2bf(a1.y); av[6]  = f2bf(a1.z); av[7]  = f2bf(a1.w);
            av[8]  = f2bf(a2.x); av[9]  = f2bf(a2.y); av[10] = f2bf(a2.z); av[11] = f2bf(a2.w);
            av[12] = f2bf(a3.x); av[13] = f2bf(a3.y); av[14] = f2bf(a3.z); av[15] = f2bf(a3.w);
            acc[mt] = __builtin_amdgcn_wmma_f32_16x16x32_bf16(
                false, av, false, bv, (short)0, acc[mt], false, false);
        }
    }

    const float bs = bias ? bias[n0 + col] : 0.f;
    #pragma unroll
    for (int mt = 0; mt < 4; ++mt) {
        const int rbase = mt * 16 + (hi ? 8 : 0);  // C/D layout: lo lanes rows 0..7, hi 8..15
        #pragma unroll
        for (int i = 0; i < 8; ++i)
            C[(size_t)(rbase + i) * N + n0 + col] = acc[mt][i] + bs;
    }
}

// ---------------------------------------------------------------------------
// Kernel 5: GRU gate combine (torch gate order r,z,n)
// ---------------------------------------------------------------------------
__global__ __launch_bounds__(256) void k_gru(const float* __restrict__ gi,
                                             const float* __restrict__ gh,
                                             const float* __restrict__ h_prev,
                                             float* __restrict__ h_new,
                                             float* __restrict__ out_h) {
    const int idx = blockIdx.x * 256 + threadIdx.x;   // BATCH*HIDDEN
    const int b = idx >> 10, h = idx & (HIDDEN - 1);
    const float* gib = gi + (size_t)b * 3 * HIDDEN;
    const float* ghb = gh + (size_t)b * 3 * HIDDEN;
    const float r = 1.f / (1.f + __expf(-(gib[h] + ghb[h])));
    const float z = 1.f / (1.f + __expf(-(gib[HIDDEN + h] + ghb[HIDDEN + h])));
    const float n = tanhf(gib[2 * HIDDEN + h] + r * ghb[2 * HIDDEN + h]);
    const float hn = (1.f - z) * n + z * h_prev[idx];
    h_new[idx] = hn;
    out_h[idx] = hn;
}

// ---------------------------------------------------------------------------
// Kernel 6: xc = [h_new, context] -> [BATCH, 2*HIDDEN]
// ---------------------------------------------------------------------------
__global__ __launch_bounds__(256) void k_build_xc(const float* __restrict__ h_new,
                                                  const float* __restrict__ ctx,
                                                  float* __restrict__ xc) {
    const int b = blockIdx.x, tid = threadIdx.x;
    float* xb = xc + (size_t)b * (2 * HIDDEN);
    for (int i = tid; i < HIDDEN; i += 256) {
        xb[i]          = h_new[(size_t)b * HIDDEN + i];
        xb[HIDDEN + i] = ctx[(size_t)b * HIDDEN + i];
    }
}

// ---------------------------------------------------------------------------
// Kernel 7: per-row max + log-sum-exp over VOCAB (logits are L2-resident)
// ---------------------------------------------------------------------------
__global__ __launch_bounds__(256) void k_lse(const float* __restrict__ logits,
                                             float* __restrict__ lm) {
    __shared__ float red[256];
    const int b = blockIdx.x, tid = threadIdx.x;
    const float* row = logits + (size_t)b * VOCAB;
    float m = -INFINITY;
    for (int i = tid; i < VOCAB; i += 256) m = fmaxf(m, row[i]);
    red[tid] = m; __syncthreads();
    for (int off = 128; off > 0; off >>= 1) {
        if (tid < off) red[tid] = fmaxf(red[tid], red[tid + off]);
        __syncthreads();
    }
    m = red[0]; __syncthreads();
    float s = 0.f;
    for (int i = tid; i < VOCAB; i += 256) s += __expf(row[i] - m);
    red[tid] = s; __syncthreads();
    for (int off = 128; off > 0; off >>= 1) {
        if (tid < off) red[tid] += red[tid + off];
        __syncthreads();
    }
    if (tid == 0) lm[b] = m + __logf(red[0]);
}

// ---------------------------------------------------------------------------
// Kernel 8: logp = logits - lm[b]  (in place on d_out)
// ---------------------------------------------------------------------------
__global__ __launch_bounds__(256) void k_logp(float* __restrict__ logits,
                                              const float* __restrict__ lm) {
    const int idx = blockIdx.x * 256 + threadIdx.x;
    if (idx >= BATCH * VOCAB) return;
    logits[idx] -= lm[idx / VOCAB];
}

// ---------------------------------------------------------------------------
// Launch
// ---------------------------------------------------------------------------
extern "C" void kernel_launch(void* const* d_in, const int* in_sizes, int n_in,
                              void* d_out, int out_size, void* d_ws, size_t ws_size,
                              hipStream_t stream) {
    const int*   last_output = (const int*)  d_in[0];
    const float* last_hidden = (const float*)d_in[1];
    const float* enc         = (const float*)d_in[2];
    const float* embedding   = (const float*)d_in[3];
    const float* attn_W      = (const float*)d_in[4];
    // d_in[5] = attn_b  (cancels under softmax; v.attn_b is constant over t)
    const float* v           = (const float*)d_in[6];
    const float* W_ih        = (const float*)d_in[7];
    const float* W_hh        = (const float*)d_in[8];
    const float* b_ih        = (const float*)d_in[9];
    const float* b_hh        = (const float*)d_in[10];
    const float* out_W       = (const float*)d_in[11];
    const float* out_b       = (const float*)d_in[12];

    float* out = (float*)d_out;                  // [64*50000 logp][64*1024 h_new]
    float* ws  = (float*)d_ws;

    // workspace layout (floats)
    float* u2   = ws;                                        // 1024
    float* ctx  = u2   + HIDDEN;                             // 64*1024
    float* x    = ctx  + BATCH * HIDDEN;                     // 64*1536
    float* gi   = x    + BATCH * (INPUT + HIDDEN);           // 64*3072
    float* gh   = gi   + BATCH * 3 * HIDDEN;                 // 64*3072
    float* hnew = gh   + BATCH * 3 * HIDDEN;                 // 64*1024
    float* xc   = hnew + BATCH * HIDDEN;                     // 64*2048
    float* lm   = xc   + BATCH * 2 * HIDDEN;                 // 64

    float* logits = out;                         // write logits straight into d_out
    float* out_h  = out + (size_t)BATCH * VOCAB;

    // 1) fold v through attn_W (hidden half cancels in softmax)
    k_attn_u2<<<HIDDEN / 256, 256, 0, stream>>>(attn_W, v, u2);
    // 2) scores + softmax + context
    k_attn_ctx<<<BATCH, 256, 0, stream>>>(enc, u2, ctx);
    // 3) x = [emb, context]
    k_build_x<<<BATCH, 256, 0, stream>>>(last_output, embedding, ctx, x);
    // 4) GRU gate GEMMs (WMMA bf16)
    k_gemm_bf16<<<(3 * HIDDEN) / 64, 128, 0, stream>>>(x, W_ih, b_ih, gi,
                                                       3 * HIDDEN, INPUT + HIDDEN);
    k_gemm_bf16<<<(3 * HIDDEN) / 64, 128, 0, stream>>>(last_hidden, W_hh, b_hh, gh,
                                                       3 * HIDDEN, HIDDEN);
    // 5) gate combine -> h_new (also written to output slot)
    k_gru<<<(BATCH * HIDDEN) / 256, 256, 0, stream>>>(gi, gh, last_hidden, hnew, out_h);
    // 6) xc = [h_new, context]
    k_build_xc<<<BATCH, 256, 0, stream>>>(hnew, ctx, xc);
    // 7) output projection (the 410 MB streaming GEMM, WMMA bf16)
    k_gemm_bf16<<<(VOCAB + 63) / 64, 128, 0, stream>>>(xc, out_W, out_b, logits,
                                                       VOCAB, 2 * HIDDEN);
    // 8) log-softmax
    k_lse<<<BATCH, 256, 0, stream>>>(logits, lm);
    k_logp<<<(BATCH * VOCAB + 255) / 256, 256, 0, stream>>>(logits, lm);
}
